// LGI_GT_63969242906723
// MI455X (gfx1250) — matrix operand order repaired and checked by reading, not compiled, not run on hardware
//
#include <hip/hip_runtime.h>

#define NNODES 16384
#define NEDGES 262144
#define DMODEL 256
#define NGRAPH 32
#define NPG    512
#define NHEAD  8
#define NLAYER 4
#define EPSV   1e-5f

typedef __attribute__((ext_vector_type(16))) _Float16 v16h;
typedef __attribute__((ext_vector_type(8)))  float    v8f;

union Frag { v16h v; unsigned u[8]; };
union U32H { unsigned u; _Float16 h[2]; };

static __device__ inline v8f vzero() {
  v8f v;
#pragma unroll
  for (int i = 0; i < 8; ++i) v[i] = 0.f;
  return v;
}

// A-fragment K index for VGPR j, lane half (16-bit A 16x32 layout, ISA 7.12.2)
static __device__ inline int a_k0(int j, int half) {
  return (j < 4) ? (2 * j + 8 * half) : (8 + 2 * j + 8 * half);
}

// low 32 bits of a flat LDS address == wave-relative LDS byte offset
static __device__ inline unsigned lds_off(const void* p) {
  return (unsigned)(unsigned long long)p;
}

// async direct memory->LDS copies (ASYNCcnt-tracked; bypass VGPRs)
static __device__ inline void async_ld_b128(unsigned ldsaddr, const void* g) {
  unsigned long long ga = (unsigned long long)g;
  asm volatile("global_load_async_to_lds_b128 %0, %1, off"
               :: "v"(ldsaddr), "v"(ga) : "memory");
}
static __device__ inline void async_ld_b32(unsigned ldsaddr, const void* g) {
  unsigned long long ga = (unsigned long long)g;
  asm volatile("global_load_async_to_lds_b32 %0, %1, off"
               :: "v"(ldsaddr), "v"(ga) : "memory");
}
static __device__ inline void wait_async0() {
  asm volatile("s_wait_asynccnt 0x0" ::: "memory");
}

// ---------------------------------------------------------------------------
// Generic WMMA GEMM: C[M,N] = act(A[M,K](f16) @ W[K,N](f32) + bias)
// Block tile 128x64, 8 waves (4x2), wave tile 32x32 = 2x2 WMMA 16x16, K step 32.
// A tile staged via async memory->LDS; W tile converted f32->f16 + transposed.
// ---------------------------------------------------------------------------
template<int RELU, int WF32, int WF16>
__global__ __launch_bounds__(256) void wmma_gemm_kernel(
    const _Float16* __restrict__ A, const float* __restrict__ W,
    const float* __restrict__ bias, float* __restrict__ Cf,
    _Float16* __restrict__ Ch, int M, int N, int K)
{
  constexpr int AST = 40;   // A tile LDS stride (halves); 80B rows keep 16B align
  constexpr int WST = 40;   // W^T tile LDS stride (halves)
  __shared__ _Float16 As[128 * AST];
  __shared__ _Float16 Wt[64 * WST];

  const int tid = threadIdx.x;
  const int lane = tid & 31, wv = tid >> 5;
  const int wm = wv >> 1, wn = wv & 1;
  const int m0 = blockIdx.x * 128, n0 = blockIdx.y * 64;
  const int half = lane >> 4, lr = lane & 15;

  v8f acc[2][2];
  acc[0][0] = vzero(); acc[0][1] = vzero();
  acc[1][0] = vzero(); acc[1][1] = vzero();

  for (int kb = 0; kb < K; kb += 32) {
    // stage A tile 128x32 f16: async global->LDS, 16B chunks
#pragma unroll
    for (int j = 0; j < 2; ++j) {
      int c = tid + j * 256;
      int row = c >> 2, cc = (c & 3) * 8;
      async_ld_b128(lds_off(&As[row * AST + cc]),
                    &A[(m0 + row) * K + kb + cc]);
    }
    // stage W tile 32x64 f32 -> transposed f16 in LDS (so B-frag K-pairs are
    // contiguous)
    {
      int idx = tid * 8;
      int kr = idx >> 6, col = idx & 63;
      const float* wp = &W[(kb + kr) * N + n0 + col];
#pragma unroll
      for (int i = 0; i < 8; ++i)
        Wt[(col + i) * WST + kr] = (_Float16)wp[i];
    }
    wait_async0();
    __syncthreads();

    Frag bfr[2];
#pragma unroll
    for (int nt = 0; nt < 2; ++nt)
#pragma unroll
      for (int j = 0; j < 8; ++j)
        bfr[nt].u[j] = *(const unsigned*)
            &Wt[(wn * 32 + nt * 16 + lr) * WST + a_k0(j, half)];

#pragma unroll
    for (int mt = 0; mt < 2; ++mt) {
      Frag af;
#pragma unroll
      for (int j = 0; j < 8; ++j)
        af.u[j] = *(const unsigned*)
            &As[(wm * 32 + mt * 16 + lr) * AST + a_k0(j, half)];
#pragma unroll
      for (int nt = 0; nt < 2; ++nt)
        acc[mt][nt] = __builtin_amdgcn_wmma_f32_16x16x32_f16(
            false, af.v, false, bfr[nt].v, (short)0, acc[mt][nt], false, false);
    }
    __syncthreads();
  }

#pragma unroll
  for (int mt = 0; mt < 2; ++mt)
#pragma unroll
    for (int nt = 0; nt < 2; ++nt) {
      int col = n0 + wn * 32 + nt * 16 + lr;
      float bv = bias[col];
#pragma unroll
      for (int j = 0; j < 8; ++j) {
        int row = m0 + wm * 32 + mt * 16 + j + 8 * half;
        float v = acc[mt][nt][j] + bv;
        if (RELU) v = fmaxf(v, 0.f);
        if (WF32) Cf[row * N + col] = v;
        if (WF16) Ch[row * N + col] = (_Float16)v;
      }
    }
}

// ---------------------------------------------------------------------------
// Flash attention per (graph, head). Head dim 32 == WMMA K dim.
// Block: 8 waves x 16 query rows = 128 queries; grid.y covers 512/128.
// K,V for the (b,h) staged in LDS; stream key blocks of 32 with online softmax.
// ---------------------------------------------------------------------------
__global__ __launch_bounds__(256) void attn_kernel(
    const _Float16* __restrict__ Q, const _Float16* __restrict__ Km,
    const _Float16* __restrict__ Vm, _Float16* __restrict__ O)
{
  constexpr int KST = 34;    // K tile row stride (halves), conflict padding
  constexpr int VST = 514;   // V^T row stride (halves)
  __shared__ _Float16 Ks[NPG * KST];       // [key][dim]
  __shared__ _Float16 Vs[32 * VST];        // [dim][key]
  __shared__ float    Ssm[8][16 * 32];     // per-wave score tile
  __shared__ _Float16 Psm[8][16 * 32];     // per-wave prob tile (f16)
  __shared__ float    Mrow[8][16], Lrow[8][16], Srow[8][16];

  const int bh = blockIdx.x;
  const int b = bh >> 3, hd = bh & 7;
  const int tid = threadIdx.x, lane = tid & 31, wv = tid >> 5;
  const int half = lane >> 4, lr = lane & 15;
  const int nodeBase = b * NPG;

  // stage K as [key][dim] via async memory->LDS (4B chunks)
  for (int i = tid; i < NPG * 16; i += 256) {
    int key = i >> 4, dp = (i & 15) * 2;
    async_ld_b32(lds_off(&Ks[key * KST + dp]),
                 &Km[(nodeBase + key) * DMODEL + hd * 32 + dp]);
  }
  // stage V transposed as [dim][key] (needs shuffle; regular path)
  for (int i = tid; i < NPG * 16; i += 256) {
    int key = i >> 4, dp = (i & 15) * 2;
    U32H t;
    t.u = *(const unsigned*)&Vm[(nodeBase + key) * DMODEL + hd * 32 + dp];
    Vs[dp * VST + key] = t.h[0];
    Vs[(dp + 1) * VST + key] = t.h[1];
  }
  if (lane < 16) { Mrow[wv][lane] = -1e30f; Lrow[wv][lane] = 0.f; }
  wait_async0();
  __syncthreads();

  // Q fragment (rows = 16 queries of this wave, K-dim = head dim 32)
  const int qrow = blockIdx.y * 128 + wv * 16;
  Frag aq;
#pragma unroll
  for (int j = 0; j < 8; ++j)
    aq.u[j] = *(const unsigned*)
        &Q[(nodeBase + qrow + lr) * DMODEL + hd * 32 + a_k0(j, half)];

  v8f o0 = vzero(), o1 = vzero();
  const float scl = 0.17677669529663687f;  // 1/sqrt(32)
  float* Sw = &Ssm[wv][0];
  _Float16* Pw = &Psm[wv][0];

  for (int kb = 0; kb < NPG; kb += 32) {
    // two score WMMAs: 16q x 16k each
#pragma unroll
    for (int s = 0; s < 2; ++s) {
      Frag bk;
#pragma unroll
      for (int j = 0; j < 8; ++j)
        bk.u[j] = *(const unsigned*)
            &Ks[(kb + s * 16 + lr) * KST + a_k0(j, half)];
      v8f sc = __builtin_amdgcn_wmma_f32_16x16x32_f16(
          false, aq.v, false, bk.v, (short)0, vzero(), false, false);
#pragma unroll
      for (int j = 0; j < 8; ++j)
        Sw[(j + 8 * half) * 32 + s * 16 + lr] = sc[j] * scl;
    }
    __syncthreads();

    // online softmax per row (lanes 0..15 own rows)
    if (lane < 16) {
      float mold = Mrow[wv][lane];
      float mx = mold;
      float* row = &Sw[lane * 32];
#pragma unroll
      for (int c = 0; c < 32; ++c) mx = fmaxf(mx, row[c]);
      float scale = __expf(mold - mx);
      float l = Lrow[wv][lane] * scale;
      _Float16* prow = &Pw[lane * 32];
#pragma unroll
      for (int c = 0; c < 32; ++c) {
        float p = __expf(row[c] - mx);
        l += p;
        prow[c] = (_Float16)p;
      }
      Mrow[wv][lane] = mx; Lrow[wv][lane] = l; Srow[wv][lane] = scale;
    }
    __syncthreads();

    // rescale running O, then O += P(16x32) @ V(32x32)
#pragma unroll
    for (int j = 0; j < 8; ++j) {
      float s = Srow[wv][j + 8 * half];
      o0[j] *= s; o1[j] *= s;
    }
    Frag ap;
#pragma unroll
    for (int j = 0; j < 8; ++j)
      ap.u[j] = *(const unsigned*)&Pw[lr * 32 + a_k0(j, half)];
    Frag bv0, bv1;
#pragma unroll
    for (int j = 0; j < 8; ++j) {
      bv0.u[j] = *(const unsigned*)&Vs[lr * VST + kb + a_k0(j, half)];
      bv1.u[j] = *(const unsigned*)&Vs[(16 + lr) * VST + kb + a_k0(j, half)];
    }
    o0 = __builtin_amdgcn_wmma_f32_16x16x32_f16(false, ap.v, false, bv0.v,
                                                (short)0, o0, false, false);
    o1 = __builtin_amdgcn_wmma_f32_16x16x32_f16(false, ap.v, false, bv1.v,
                                                (short)0, o1, false, false);
    __syncthreads();
  }

#pragma unroll
  for (int j = 0; j < 8; ++j) {
    float inv = 1.f / Lrow[wv][j + 8 * half];
    int row = nodeBase + qrow + j + 8 * half;
    O[row * DMODEL + hd * 32 + lr]      = (_Float16)(o0[j] * inv);
    O[row * DMODEL + hd * 32 + 16 + lr] = (_Float16)(o1[j] * inv);
  }
}

// ---------------------------------------------------------------------------
// Plumbing kernels (bandwidth-trivial vs 192MB L2; fp32 VALU)
// ---------------------------------------------------------------------------
__global__ void zero_kernel(float* p) {
  p[blockIdx.x * 256 + threadIdx.x] = 0.f;
}

// per-channel sum/sumsq of (A [+ Bv]) over n rows; one block per channel
__global__ void stats_kernel(const float* __restrict__ A,
                             const float* __restrict__ Bv,
                             float* __restrict__ stats, int n, int C) {
  int c = blockIdx.x, tid = threadIdx.x;
  __shared__ float s1[256], s2[256];
  float a1 = 0.f, a2 = 0.f;
  for (int i = tid; i < n; i += 256) {
    float v = A[i * C + c];
    if (Bv) v += Bv[i * C + c];
    a1 += v; a2 += v * v;
  }
  s1[tid] = a1; s2[tid] = a2; __syncthreads();
  for (int o = 128; o; o >>= 1) {
    if (tid < o) { s1[tid] += s1[tid + o]; s2[tid] += s2[tid + o]; }
    __syncthreads();
  }
  if (!tid) { stats[c] = s1[0]; stats[C + c] = s2[0]; }
}

// node encoder: h = [node_emb[x] , Lin(BN(rw))]
__global__ void encoder_kernel(const int* __restrict__ x,
    const float* __restrict__ rw, const float* __restrict__ node_emb,
    const float* __restrict__ rwg, const float* __restrict__ rwb,
    const float* __restrict__ Wp, const float* __restrict__ bp,
    const float* __restrict__ rwst, float* __restrict__ h) {
  int n = blockIdx.x, c = threadIdx.x;
  float v;
  if (c < 224) {
    v = node_emb[x[n] * 224 + c];
  } else {
    int i = c - 224;
    float acc = bp[i];
    for (int j = 0; j < 20; ++j) {
      float mean = rwst[j] * (1.f / NNODES);
      float var  = rwst[20 + j] * (1.f / NNODES) - mean * mean;
      float pe = (rw[n * 20 + j] - mean) * rsqrtf(var + EPSV) * rwg[j] + rwb[j];
      acc += pe * Wp[j * 32 + i];
    }
    v = acc;
  }
  h[n * DMODEL + c] = v;
}

// GINE message + scatter-add: agg[dst] += relu(h[src] + e)
__global__ void gine_scatter_kernel(const int* __restrict__ src,
    const int* __restrict__ dst, const int* __restrict__ eattr,
    const float* __restrict__ edge_emb, const float* __restrict__ h,
    float* __restrict__ agg) {
  int idx = blockIdx.x * 256 + threadIdx.x;
  int e = idx >> 6;
  int c = (idx & 63) * 4;
  int s = src[e], d = dst[e], a = eattr[e];
  float4 hv = *(const float4*)&h[s * DMODEL + c];
  float4 ev = *(const float4*)&edge_emb[a * DMODEL + c];
  atomicAdd(&agg[d * DMODEL + c + 0], fmaxf(hv.x + ev.x, 0.f));
  atomicAdd(&agg[d * DMODEL + c + 1], fmaxf(hv.y + ev.y, 0.f));
  atomicAdd(&agg[d * DMODEL + c + 2], fmaxf(hv.z + ev.z, 0.f));
  atomicAdd(&agg[d * DMODEL + c + 3], fmaxf(hv.w + ev.w, 0.f));
}

// xin(f16) = h + agg
__global__ void add_to_h16_kernel(const float* __restrict__ a,
    const float* __restrict__ b, _Float16* __restrict__ o) {
  int i = blockIdx.x * 256 + threadIdx.x;
  o[i] = (_Float16)(a[i] + b[i]);
}

// h = BN(h + g); also emit f16 copy
__global__ void bn_apply_kernel(float* __restrict__ h,
    const float* __restrict__ g, const float* __restrict__ stats,
    const float* __restrict__ gamma, const float* __restrict__ beta,
    _Float16* __restrict__ xh) {
  int idx = blockIdx.x * 256 + threadIdx.x;
  int c = idx & (DMODEL - 1);
  float mean = stats[c] * (1.f / NNODES);
  float var  = stats[DMODEL + c] * (1.f / NNODES) - mean * mean;
  float v = (h[idx] + g[idx] - mean) * rsqrtf(var + EPSV) * gamma[c] + beta[c];
  h[idx] = v;
  xh[idx] = (_Float16)v;
}

// t = LN(a + b); one block per node
__global__ void ln_kernel(const float* __restrict__ A,
    const float* __restrict__ Bv, const float* __restrict__ gamma,
    const float* __restrict__ beta, float* __restrict__ outF,
    _Float16* __restrict__ outH) {
  int n = blockIdx.x, c = threadIdx.x;
  __shared__ float s1[256], s2[256];
  float v = A[n * DMODEL + c] + Bv[n * DMODEL + c];
  s1[c] = v; s2[c] = v * v; __syncthreads();
  for (int o = 128; o; o >>= 1) {
    if (c < o) { s1[c] += s1[c + o]; s2[c] += s2[c + o]; }
    __syncthreads();
  }
  float mean = s1[0] * (1.f / DMODEL);
  float var  = s2[0] * (1.f / DMODEL) - mean * mean;
  float r = (v - mean) * rsqrtf(var + EPSV) * gamma[c] + beta[c];
  outF[n * DMODEL + c] = r;
  outH[n * DMODEL + c] = (_Float16)r;
}

// h += LN(t + f)   (outer residual; h==out invariant)
__global__ void ln_acc_kernel(const float* __restrict__ T,
    const float* __restrict__ F, const float* __restrict__ gamma,
    const float* __restrict__ beta, float* __restrict__ h) {
  int n = blockIdx.x, c = threadIdx.x;
  __shared__ float s1[256], s2[256];
  float v = T[n * DMODEL + c] + F[n * DMODEL + c];
  s1[c] = v; s2[c] = v * v; __syncthreads();
  for (int o = 128; o; o >>= 1) {
    if (c < o) { s1[c] += s1[c + o]; s2[c] += s2[c + o]; }
    __syncthreads();
  }
  float mean = s1[0] * (1.f / DMODEL);
  float var  = s2[0] * (1.f / DMODEL) - mean * mean;
  float r = (v - mean) * rsqrtf(var + EPSV) * gamma[c] + beta[c];
  h[n * DMODEL + c] += r;
}

// mean pool per graph (equal-sized graphs)
__global__ void readout_kernel(const float* __restrict__ h,
                               float* __restrict__ pooled) {
  int b = blockIdx.x, c = threadIdx.x;
  float s = 0.f;
  for (int i = 0; i < NPG; ++i) s += h[(b * NPG + i) * DMODEL + c];
  pooled[b * DMODEL + c] = s * (1.f / NPG);
}

// tiny head GEMMs: one wave per output element
__global__ void dot_kernel(const float* __restrict__ A,
    const float* __restrict__ W, const float* __restrict__ bias,
    float* __restrict__ out, int M, int N, int K, int relu) {
  int o = blockIdx.x * 8 + (threadIdx.x >> 5);
  int lane = threadIdx.x & 31;
  if (o >= M * N) return;
  int m = o / N, n = o - m * N;
  float acc = 0.f;
  for (int k = lane; k < K; k += 32) {
    if (k + 256 < K) __builtin_prefetch(&W[(k + 256) * N + n], 0, 0);
    acc += A[m * K + k] * W[k * N + n];
  }
  for (int off = 16; off; off >>= 1) acc += __shfl_xor(acc, off);
  if (!lane) {
    float v = acc + bias[n];
    if (relu) v = fmaxf(v, 0.f);
    out[m * N + n] = v;
  }
}

// ---------------------------------------------------------------------------
extern "C" void kernel_launch(void* const* d_in, const int* in_sizes, int n_in,
                              void* d_out, int out_size, void* d_ws,
                              size_t ws_size, hipStream_t stream) {
  (void)in_sizes; (void)n_in; (void)out_size; (void)ws_size;
  auto F = [&](int i) { return (const float*)d_in[i]; };

  // top-level inputs (insertion order), then params pytree leaves
  // (dict keys alphabetical per JAX flatten; tuples/lists in order)
  const int*   x_idx   = (const int*)d_in[0];
  const float* rw      = F(1);
  const int*   eidx    = (const int*)d_in[2];
  const int*   eattr   = (const int*)d_in[3];
  const float* edgeE   = F(5);
  const float* hW0 = F(6);  const float* hb0 = F(7);
  const float* hW1 = F(8);  const float* hb1 = F(9);
  const float* hW2 = F(10); const float* hb2 = F(11);
  // layers at 12 + 22*l: attn(Wq,bq,Wk,bk,Wv,bv,Wo,bo), bn_b, bn_g,
  //                      ffn(Wf1,bf1,Wf2,bf2), gin(Wg1,bg1,Wg2,bg2),
  //                      ln1_b, ln1_g, ln2_b, ln2_g
  const float* nodeE = F(100);
  const float* rwb   = F(101);
  const float* rwg   = F(102);
  const float* rwWp  = F(103);
  const float* rwbp  = F(104);

  const int* src = eidx;
  const int* dst = eidx + NEDGES;

  // workspace layout
  char* w = (char*)d_ws;
  const size_t ND = (size_t)NNODES * DMODEL;
  float* h    = (float*)w; w += ND * 4;
  float* g    = (float*)w; w += ND * 4;
  float* t    = (float*)w; w += ND * 4;
  float* agg  = (float*)w; w += ND * 4;
  float* stats  = (float*)w; w += 2 * DMODEL * 4;
  float* rwst   = (float*)w; w += 64 * 4;
  float* pooled = (float*)w; w += NGRAPH * DMODEL * 4;
  float* hh1  = (float*)w; w += NGRAPH * 2048 * 4;
  float* hh2  = (float*)w; w += NGRAPH * 2048 * 4;
  w = (char*)(((size_t)w + 255) & ~(size_t)255);
  _Float16* hp0 = (_Float16*)w;
  _Float16* hp1 = hp0 + ND;
  _Float16* hp2 = hp1 + ND;
  _Float16* hp3 = hp2 + ND;   // hp2..hp3 together hold the 2D FFN hidden

  const dim3 blk(256);
  const dim3 gEW(NNODES * DMODEL / 256);     // elementwise over [NN,D]
  const dim3 gGemm256(NNODES / 128, DMODEL / 64);
  const dim3 gGemm512(NNODES / 128, 512 / 64);
  const dim3 gAttn(NGRAPH * NHEAD, NPG / 128);

  // ---- encoder ----
  stats_kernel<<<20, blk, 0, stream>>>(rw, nullptr, rwst, NNODES, 20);
  encoder_kernel<<<NNODES, blk, 0, stream>>>(x_idx, rw, nodeE, rwg, rwb,
                                             rwWp, rwbp, rwst, h);

  for (int l = 0; l < NLAYER; ++l) {
    const int base = 12 + 22 * l;
    const float* Wq = F(base + 0),  *bq = F(base + 1);
    const float* Wk = F(base + 2),  *bk = F(base + 3);
    const float* Wv = F(base + 4),  *bv = F(base + 5);
    const float* Wo = F(base + 6),  *bo = F(base + 7);
    const float* bnb = F(base + 8), *bng = F(base + 9);
    const float* Wf1 = F(base + 10), *bf1 = F(base + 11);
    const float* Wf2 = F(base + 12), *bf2 = F(base + 13);
    const float* Wg1 = F(base + 14), *bg1 = F(base + 15);
    const float* Wg2 = F(base + 16), *bg2 = F(base + 17);
    const float* l1b = F(base + 18), *l1g = F(base + 19);
    const float* l2b = F(base + 20), *l2g = F(base + 21);

    // GINE conv
    zero_kernel<<<gEW, blk, 0, stream>>>(agg);
    gine_scatter_kernel<<<NEDGES * 64 / 256, blk, 0, stream>>>(
        src, dst, eattr, edgeE, h, agg);
    add_to_h16_kernel<<<gEW, blk, 0, stream>>>(h, agg, hp0);
    wmma_gemm_kernel<1, 0, 1><<<gGemm256, blk, 0, stream>>>(
        hp0, Wg1, bg1, nullptr, hp1, NNODES, DMODEL, DMODEL);
    wmma_gemm_kernel<0, 1, 0><<<gGemm256, blk, 0, stream>>>(
        hp1, Wg2, bg2, g, nullptr, NNODES, DMODEL, DMODEL);
    stats_kernel<<<DMODEL, blk, 0, stream>>>(h, g, stats, NNODES, DMODEL);
    bn_apply_kernel<<<gEW, blk, 0, stream>>>(h, g, stats, bng, bnb, hp0);

    // transformer: QKV proj -> flash attention -> Wo
    wmma_gemm_kernel<0, 0, 1><<<gGemm256, blk, 0, stream>>>(
        hp0, Wq, bq, nullptr, hp1, NNODES, DMODEL, DMODEL);
    wmma_gemm_kernel<0, 0, 1><<<gGemm256, blk, 0, stream>>>(
        hp0, Wk, bk, nullptr, hp2, NNODES, DMODEL, DMODEL);
    wmma_gemm_kernel<0, 0, 1><<<gGemm256, blk, 0, stream>>>(
        hp0, Wv, bv, nullptr, hp3, NNODES, DMODEL, DMODEL);
    attn_kernel<<<gAttn, blk, 0, stream>>>(hp1, hp2, hp3, hp0);
    wmma_gemm_kernel<0, 1, 0><<<gGemm256, blk, 0, stream>>>(
        hp0, Wo, bo, g, nullptr, NNODES, DMODEL, DMODEL);
    ln_kernel<<<NNODES, blk, 0, stream>>>(h, g, l1g, l1b, t, hp1);

    // FFN + second LN + outer residual (h == out invariant)
    wmma_gemm_kernel<1, 0, 1><<<gGemm512, blk, 0, stream>>>(
        hp1, Wf1, bf1, nullptr, hp2, NNODES, 512, DMODEL);
    wmma_gemm_kernel<0, 1, 0><<<gGemm256, blk, 0, stream>>>(
        hp2, Wf2, bf2, g, nullptr, NNODES, DMODEL, 512);
    ln_acc_kernel<<<NNODES, blk, 0, stream>>>(t, g, l2g, l2b, h);
  }

  // readout + MLP head
  readout_kernel<<<NGRAPH, blk, 0, stream>>>(h, pooled);
  dot_kernel<<<NGRAPH * 2048 / 8, blk, 0, stream>>>(
      pooled, hW0, hb0, hh1, NGRAPH, 2048, DMODEL, 1);
  dot_kernel<<<NGRAPH * 2048 / 8, blk, 0, stream>>>(
      hh1, hW1, hb1, hh2, NGRAPH, 2048, 2048, 1);
  dot_kernel<<<(NGRAPH + 7) / 8, blk, 0, stream>>>(
      hh2, hW2, hb2, (float*)d_out, NGRAPH, 1, 2048, 0);
}
